// Encoder_37718402794103
// MI455X (gfx1250) — compile-verified, matrix operand order checked by reading, not run on hardware
//
#include <hip/hip_runtime.h>
#include <hip/hip_bf16.h>
#include <math.h>

// ---------------- problem constants (from reference) ----------------
#define NN    2048            // nodes
#define BB    512             // batch
#define TT    9               // IN_DIM
#define GC    (BB * TT)       // 4608 GEMM columns
#define HID   64
#define HEADS 2
#define FD    (HEADS * HID)   // 128
#define CB    64              // batch chunk for conv1 scratch
#define NEG_SLOPE 0.2f

#define KTILES (NN / 32)      // 64   WMMA K-tiles
#define MTILES (NN / 16)      // 128  A row tiles
#define NTILES (GC / 16)      // 288  B col tiles

typedef __attribute__((ext_vector_type(16))) __bf16 v16bf;
typedef __attribute__((ext_vector_type(8)))  float  v8f;
typedef __attribute__((ext_vector_type(4)))  int    v4i;

// CDNA5 async global->LDS path (guarded: falls back to sync copy if the
// builtin names are absent in this toolchain)
#if defined(__HIP_DEVICE_COMPILE__) && defined(__gfx1250__) && \
    __has_builtin(__builtin_amdgcn_global_load_async_to_lds_b128) && \
    __has_builtin(__builtin_amdgcn_s_wait_asynccnt)
#define USE_ASYNC_LDS 1
typedef __attribute__((address_space(1))) v4i* gvp_t;   // global v4i*
typedef __attribute__((address_space(3))) v4i* lvp_t;   // LDS v4i*
#else
#define USE_ASYNC_LDS 0
#endif

// ---------------- helpers ----------------
__device__ __forceinline__ void atomicMaxF(float* addr, float val) {
  if (val >= 0.0f) atomicMax((int*)addr, __float_as_int(val));
  else             atomicMin((unsigned int*)addr, __float_as_uint(val));
}

__device__ __forceinline__ float fast_tanh(float x) {
  x = fminf(9.0f, fmaxf(-9.0f, x));
  float e = __expf(2.0f * x);
  return (e - 1.0f) / (e + 1.0f);
}

__global__ void fill_f32(float* __restrict__ p, float v, int n) {
  int i = blockIdx.x * blockDim.x + threadIdx.x;
  if (i < n) p[i] = v;
}

// fragment k-offset for 16-bit WMMA A/B tiles: lane<16 holds k {0..7,16..23},
// lane>=16 holds k {8..15,24..31}; element i -> koff
__device__ __forceinline__ int frag_koff(int lane, int i) {
  return (i < 8 ? i : i + 8) + ((lane >= 16) ? 8 : 0);
}

// Wf[kt][mt][lane][16] = bf16 fragment-ordered W (A-matrix for WMMA)
__global__ __launch_bounds__(256) void build_wfrag(const float* __restrict__ W,
                                                   v16bf* __restrict__ Wf) {
  int idx = blockIdx.x * blockDim.x + threadIdx.x;      // kt*4096 + mt*32 + lane
  if (idx >= KTILES * MTILES * 32) return;
  int kt = idx >> 12;
  int mt = (idx >> 5) & (MTILES - 1);
  int lane = idx & 31;
  int m = mt * 16 + (lane & 15);
  v16bf v;
  #pragma unroll
  for (int i = 0; i < 16; ++i) {
    int k = kt * 32 + frag_koff(lane, i);
    v[i] = (__bf16)W[(size_t)m * NN + k];
  }
  Wf[idx] = v;
}

// Xf[kt][nt][lane][16] = bf16 fragment-ordered Xmat (B-matrix), where
// Xmat[k][c] = input[b][t][k] (c = b*9+t), i.e. input flat index c*NN + k
__global__ __launch_bounds__(256) void build_xfrag(const float* __restrict__ in,
                                                   v16bf* __restrict__ Xf) {
  int idx = blockIdx.x * blockDim.x + threadIdx.x;      // kt*9216 + nt*32 + lane
  if (idx >= KTILES * NTILES * 32) return;
  int kt = idx / (NTILES * 32);
  int rem = idx - kt * (NTILES * 32);
  int nt = rem >> 5;
  int lane = rem & 31;
  int c = nt * 16 + (lane & 15);
  v16bf v;
  #pragma unroll
  for (int i = 0; i < 16; ++i) {
    int k = kt * 32 + frag_koff(lane, i);
    v[i] = (__bf16)in[(size_t)c * NN + k];
  }
  Xf[idx] = v;
}

// ael[h][t] = sum_d al1[h][d]*fc1_w[h*64+d][t]   (collapses el/er to 9-dot)
__global__ void build_ael(const float* __restrict__ al1, const float* __restrict__ ar1,
                          const float* __restrict__ fc1w,
                          float* __restrict__ ael, float* __restrict__ aer) {
  int i = threadIdx.x;
  if (i < HEADS * TT) {
    int h = i / TT, t = i % TT;
    float sl = 0.f, sr = 0.f;
    for (int d = 0; d < HID; ++d) {
      float w = fc1w[(h * HID + d) * TT + t];
      sl += al1[h * HID + d] * w;
      sr += ar1[h * HID + d] * w;
    }
    ael[i] = sl; aer[i] = sr;
  }
}

// ---------------- kernel 1: Y = tanh(W @ Xmat), bf16 WMMA, f32 accumulate ---
// Block tile 128x128 (8 mtiles x 8 ntiles), BK=32, double-buffered LDS,
// fragment-native LDS layout: every fragment load = 32 contiguous bytes.
__global__ __launch_bounds__(256) void gemm_tanh_wmma(const v16bf* __restrict__ Wf,
                                                      const v16bf* __restrict__ Xf,
                                                      float* __restrict__ Y) {
  __shared__ v16bf As[2][8][32];    // 8 KB per buffer
  __shared__ v16bf Bs[2][8][32];

  const int tid  = threadIdx.x;
  const int lane = tid & 31;
  const int wid  = tid >> 5;                 // 8 waves
  const int m0   = blockIdx.y * 8;           // mtile base
  const int n0   = blockIdx.x * 8;           // ntile base
  const int wmi  = (wid & 3) * 2;            // wave's 2 mtiles
  const int wni  = (wid >> 2) * 4;           // wave's 4 ntiles

  v8f acc[2][4];
  #pragma unroll
  for (int tm = 0; tm < 2; ++tm)
    #pragma unroll
    for (int tn = 0; tn < 4; ++tn)
      #pragma unroll
      for (int rr = 0; rr < 8; ++rr) acc[tm][tn][rr] = 0.0f;

  auto stage = [&](int kt, int buf) {
    const v16bf* gA = Wf + ((size_t)kt * MTILES + m0) * 32;   // 8 KB contiguous
    const v16bf* gB = Xf + ((size_t)kt * NTILES + n0) * 32;   // 8 KB contiguous
    v16bf* lA = &As[buf][0][0];
    v16bf* lB = &Bs[buf][0][0];
#if USE_ASYNC_LDS
    {
      char* la = (char*)(lA + tid);
      char* lb = (char*)(lB + tid);
      const char* ga = (const char*)(gA + tid);
      const char* gb = (const char*)(gB + tid);
      __builtin_amdgcn_global_load_async_to_lds_b128((gvp_t)(ga),      (lvp_t)(la),      0, 0);
      __builtin_amdgcn_global_load_async_to_lds_b128((gvp_t)(ga + 16), (lvp_t)(la + 16), 0, 0);
      __builtin_amdgcn_global_load_async_to_lds_b128((gvp_t)(gb),      (lvp_t)(lb),      0, 0);
      __builtin_amdgcn_global_load_async_to_lds_b128((gvp_t)(gb + 16), (lvp_t)(lb + 16), 0, 0);
    }
#else
    lA[tid] = gA[tid];
    lB[tid] = gB[tid];
#endif
    if (kt + 1 < KTILES) {   // prefetch tile after next into L2
      __builtin_prefetch(gA + MTILES * 32 + tid, 0, 3);
      __builtin_prefetch(gB + NTILES * 32 + tid, 0, 3);
    }
  };

  stage(0, 0);
#if USE_ASYNC_LDS
  __builtin_amdgcn_s_wait_asynccnt(0);
#endif
  __syncthreads();

  for (int kt = 0; kt < KTILES; ++kt) {
    const int buf = kt & 1;
    if (kt + 1 < KTILES) stage(kt + 1, buf ^ 1);

    v16bf afrag[2], bfrag[4];
    #pragma unroll
    for (int tm = 0; tm < 2; ++tm) afrag[tm] = As[buf][wmi + tm][lane];
    #pragma unroll
    for (int tn = 0; tn < 4; ++tn) bfrag[tn] = Bs[buf][wni + tn][lane];

    #pragma unroll
    for (int tm = 0; tm < 2; ++tm)
      #pragma unroll
      for (int tn = 0; tn < 4; ++tn)
        acc[tm][tn] = __builtin_amdgcn_wmma_f32_16x16x32_bf16(
            false, afrag[tm], false, bfrag[tn], (short)0, acc[tm][tn], false, false);

    if (kt + 1 < KTILES) {
#if USE_ASYNC_LDS
      __builtin_amdgcn_s_wait_asynccnt(0);
#endif
      __syncthreads();
    }
  }

  // epilogue: tanh + store (C/D layout: elem rr -> row rr + 8*(lane>=16), col lane&15)
  #pragma unroll
  for (int tm = 0; tm < 2; ++tm)
    #pragma unroll
    for (int tn = 0; tn < 4; ++tn)
      #pragma unroll
      for (int rr = 0; rr < 8; ++rr) {
        int row = (m0 + wmi + tm) * 16 + rr + 8 * (lane >> 4);
        int col = (n0 + wni + tn) * 16 + (lane & 15);
        Y[(size_t)row * GC + col] = fast_tanh(acc[tm][tn][rr]);
      }
}

// ---------------- conv1: node-level el/er (collapsed) ----------------
__global__ void node1(const float* __restrict__ Y, const float* __restrict__ ael,
                      const float* __restrict__ aer, float* __restrict__ el,
                      float* __restrict__ er, int b0) {
  int idx = blockIdx.x * blockDim.x + threadIdx.x;       // cb*NN + n
  if (idx >= CB * NN) return;
  int cb = idx >> 11, n = idx & (NN - 1);
  const float* yp = Y + (size_t)n * GC + (size_t)(b0 + cb) * TT;
  float y[TT];
  #pragma unroll
  for (int t = 0; t < TT; ++t) y[t] = yp[t];
  #pragma unroll
  for (int h = 0; h < HEADS; ++h) {
    float sl = 0.f, sr = 0.f;
    #pragma unroll
    for (int t = 0; t < TT; ++t) { sl += y[t] * ael[h * TT + t]; sr += y[t] * aer[h * TT + t]; }
    el[idx * HEADS + h] = sl;
    er[idx * HEADS + h] = sr;
  }
}

// ---------------- conv1: per-dst softmax max ----------------
__global__ void edge_max1(const int* __restrict__ src, const int* __restrict__ dst,
                          const float* __restrict__ el, const float* __restrict__ er,
                          float* __restrict__ m, int E) {
  int idx = blockIdx.x * blockDim.x + threadIdx.x;
  if (idx >= E * CB) return;
  int e = idx % E, cb = idx / E;
  int s = src[e], d = dst[e];
  #pragma unroll
  for (int h = 0; h < HEADS; ++h) {
    float v = el[(cb * NN + s) * HEADS + h] + er[(cb * NN + d) * HEADS + h];
    v = v > 0.f ? v : NEG_SLOPE * v;
    atomicMaxF(&m[(cb * NN + d) * HEADS + h], v);
  }
}

// ---------------- conv1: exp-sum + weighted numerator (feat recomputed) ----
__global__ __launch_bounds__(256) void edge_acc1(const int* __restrict__ src,
    const int* __restrict__ dst, const float* __restrict__ Y,
    const float* __restrict__ fc1w, const float* __restrict__ el,
    const float* __restrict__ er, const float* __restrict__ m,
    float* __restrict__ ssum, float* __restrict__ num, int b0, int E) {
  __shared__ float wsh[FD * TT];
  for (int i = threadIdx.x; i < FD * TT; i += 256) wsh[i] = fc1w[i];
  __syncthreads();
  int idx = blockIdx.x * 256 + threadIdx.x;
  if (idx >= E * CB) return;
  int e = idx % E, cb = idx / E;
  int s = src[e], d = dst[e];
  float ex[HEADS];
  #pragma unroll
  for (int h = 0; h < HEADS; ++h) {
    float v = el[(cb * NN + s) * HEADS + h] + er[(cb * NN + d) * HEADS + h];
    v = v > 0.f ? v : NEG_SLOPE * v;
    ex[h] = __expf(v - m[(cb * NN + d) * HEADS + h]);
    atomicAdd(&ssum[(cb * NN + d) * HEADS + h], ex[h]);
  }
  const float* yp = Y + (size_t)s * GC + (size_t)(b0 + cb) * TT;
  float y[TT];
  #pragma unroll
  for (int t = 0; t < TT; ++t) y[t] = yp[t];
  float* np = num + ((size_t)cb * NN + d) * FD;
  for (int c = 0; c < FD; ++c) {
    float f = 0.f;
    #pragma unroll
    for (int t = 0; t < TT; ++t) f += y[t] * wsh[c * TT + t];
    atomicAdd(&np[c], f * ex[c >> 6]);
  }
}

// ---------------- conv1 finalize: elu(num/s + b1) -> feat2 (fc2 contraction)
__global__ void finalize1(const float* __restrict__ num, const float* __restrict__ ssum,
                          const float* __restrict__ fc1b, const float* __restrict__ fc2w,
                          float* __restrict__ feat2, int b0) {
  int idx = blockIdx.x * blockDim.x + threadIdx.x;       // cb*NN + n
  if (idx >= CB * NN) return;
  int cb = idx >> 11, n = idx & (NN - 1);
  const float* np = num + (size_t)idx * FD;
  float s0 = ssum[idx * HEADS + 0], s1 = ssum[idx * HEADS + 1];
  float acc = 0.f;
  for (int c = 0; c < FD; ++c) {
    float v = np[c] / (c < HID ? s0 : s1) + fc1b[c];
    v = v > 0.f ? v : (__expf(v) - 1.0f);                // elu
    acc += v * fc2w[c];
  }
  feat2[(size_t)(b0 + cb) * NN + n] = acc;
}

// ---------------- conv2 (1 head, 1 dim) ----------------
__global__ void edge_max2(const int* __restrict__ src, const int* __restrict__ dst,
                          const float* __restrict__ feat2, const float* __restrict__ al2,
                          const float* __restrict__ ar2, float* __restrict__ m2, int E) {
  int idx = blockIdx.x * blockDim.x + threadIdx.x;
  if (idx >= E * BB) return;
  int e = idx % E, b = idx / E;
  float v = al2[0] * feat2[(size_t)b * NN + src[e]] +
            ar2[0] * feat2[(size_t)b * NN + dst[e]];
  v = v > 0.f ? v : NEG_SLOPE * v;
  atomicMaxF(&m2[(size_t)b * NN + dst[e]], v);
}

__global__ void edge_acc2(const int* __restrict__ src, const int* __restrict__ dst,
                          const float* __restrict__ feat2, const float* __restrict__ al2,
                          const float* __restrict__ ar2, const float* __restrict__ m2,
                          float* __restrict__ s2, float* __restrict__ num2, int E) {
  int idx = blockIdx.x * blockDim.x + threadIdx.x;
  if (idx >= E * BB) return;
  int e = idx % E, b = idx / E;
  size_t bs = (size_t)b * NN + src[e];
  size_t bd = (size_t)b * NN + dst[e];
  float v = al2[0] * feat2[bs] + ar2[0] * feat2[bd];
  v = v > 0.f ? v : NEG_SLOPE * v;
  float ex = __expf(v - m2[bd]);
  atomicAdd(&s2[bd], ex);
  atomicAdd(&num2[bd], feat2[bs] * ex);
}

__global__ void finalize2(const float* __restrict__ num2, const float* __restrict__ s2,
                          const float* __restrict__ fc2b, float* __restrict__ h2, int n) {
  int i = blockIdx.x * blockDim.x + threadIdx.x;
  if (i < n) h2[i] = num2[i] / s2[i] + fc2b[0];
}

// ---------------- final: out[b] = sum_n h2[b,n]*lin_w[n] + lin_b ------------
__global__ __launch_bounds__(256) void final_lin(const float* __restrict__ h2,
                                                 const float* __restrict__ linw,
                                                 const float* __restrict__ linb,
                                                 float* __restrict__ out) {
  __shared__ float red[256];
  int b = blockIdx.x;
  float acc = 0.f;
  for (int n = threadIdx.x; n < NN; n += 256)
    acc += h2[(size_t)b * NN + n] * linw[n];
  red[threadIdx.x] = acc;
  __syncthreads();
  for (int s = 128; s > 0; s >>= 1) {
    if (threadIdx.x < s) red[threadIdx.x] += red[threadIdx.x + s];
    __syncthreads();
  }
  if (threadIdx.x == 0) out[b] = red[0] + linb[0];
}

// ---------------- launcher ----------------
extern "C" void kernel_launch(void* const* d_in, const int* in_sizes, int n_in,
                              void* d_out, int out_size, void* d_ws, size_t ws_size,
                              hipStream_t stream) {
  const float* input = (const float*)d_in[0];
  const float* W     = (const float*)d_in[1];
  const float* fc1_w = (const float*)d_in[2];
  const float* fc1_b = (const float*)d_in[3];
  const float* al1   = (const float*)d_in[4];
  const float* ar1   = (const float*)d_in[5];
  const float* fc2_w = (const float*)d_in[6];
  const float* fc2_b = (const float*)d_in[7];
  const float* al2   = (const float*)d_in[8];
  const float* ar2   = (const float*)d_in[9];
  const float* lin_w = (const float*)d_in[10];
  const float* lin_b = (const float*)d_in[11];
  const int*   src   = (const int*)d_in[12];
  const int*   dst   = (const int*)d_in[13];
  const int E = in_sizes[12];

  char* ws = (char*)d_ws;
  size_t off = 0;
  auto alloc = [&](size_t bytes) { size_t o = off; off += (bytes + 255) & ~(size_t)255; return o; };

  v16bf* Wf    = (v16bf*)(ws + alloc((size_t)NN * NN * 2));
  v16bf* Xf    = (v16bf*)(ws + alloc((size_t)NN * GC * 2));
  float* Y     = (float*)(ws + alloc((size_t)NN * GC * 4));
  float* ael   = (float*)(ws + alloc(HEADS * TT * 4));
  float* aer   = (float*)(ws + alloc(HEADS * TT * 4));
  float* feat2 = (float*)(ws + alloc((size_t)BB * NN * 4));
  float* h2    = (float*)(ws + alloc((size_t)BB * NN * 4));
  float* m2    = (float*)(ws + alloc((size_t)BB * NN * 4));
  float* s2    = (float*)(ws + alloc((size_t)BB * NN * 4));
  float* num2  = (float*)(ws + alloc((size_t)BB * NN * 4));
  float* el1   = (float*)(ws + alloc((size_t)CB * NN * HEADS * 4));
  float* er1   = (float*)(ws + alloc((size_t)CB * NN * HEADS * 4));
  float* m1    = (float*)(ws + alloc((size_t)CB * NN * HEADS * 4));
  float* s1    = (float*)(ws + alloc((size_t)CB * NN * HEADS * 4));
  float* num1  = (float*)(ws + alloc((size_t)CB * NN * FD * 4));
  (void)ws_size; (void)n_in;

  // 1) precision/layout prep: fragment-native bf16 operands
  build_wfrag<<<(KTILES * MTILES * 32 + 255) / 256, 256, 0, stream>>>(W, Wf);
  build_xfrag<<<(KTILES * NTILES * 32 + 255) / 256, 256, 0, stream>>>(input, Xf);
  build_ael<<<1, 32, 0, stream>>>(al1, ar1, fc1_w, ael, aer);

  // 2) dominant GEMM: Y = tanh(W @ Xmat), bf16 WMMA, async-DMA staged
  gemm_tanh_wmma<<<dim3(NTILES / 8, MTILES / 8), 256, 0, stream>>>(Wf, Xf, Y);

  // 3) conv2 accumulator init (full batch)
  fill_f32<<<(BB * NN + 255) / 256, 256, 0, stream>>>(m2, -1e30f, BB * NN);
  (void)hipMemsetAsync(s2, 0, (size_t)BB * NN * 4, stream);
  (void)hipMemsetAsync(num2, 0, (size_t)BB * NN * 4, stream);

  // 4) conv1, chunked over batch
  const int eg1 = (E * CB + 255) / 256;
  for (int b0 = 0; b0 < BB; b0 += CB) {
    node1<<<(CB * NN + 255) / 256, 256, 0, stream>>>(Y, ael, aer, el1, er1, b0);
    fill_f32<<<(CB * NN * HEADS + 255) / 256, 256, 0, stream>>>(m1, -1e30f, CB * NN * HEADS);
    (void)hipMemsetAsync(s1, 0, (size_t)CB * NN * HEADS * 4, stream);
    (void)hipMemsetAsync(num1, 0, (size_t)CB * NN * FD * 4, stream);
    edge_max1<<<eg1, 256, 0, stream>>>(src, dst, el1, er1, m1, E);
    edge_acc1<<<eg1, 256, 0, stream>>>(src, dst, Y, fc1_w, el1, er1, m1, s1, num1, b0, E);
    finalize1<<<(CB * NN + 255) / 256, 256, 0, stream>>>(num1, s1, fc1_b, fc2_w, feat2, b0);
  }

  // 5) conv2 over full batch
  const int eg2 = (E * BB + 255) / 256;
  edge_max2<<<eg2, 256, 0, stream>>>(src, dst, feat2, al2, ar2, m2, E);
  edge_acc2<<<eg2, 256, 0, stream>>>(src, dst, feat2, al2, ar2, m2, s2, num2, E);
  finalize2<<<(BB * NN + 255) / 256, 256, 0, stream>>>(num2, s2, fc2_b, h2, BB * NN);

  // 6) output projection
  final_lin<<<BB, 256, 0, stream>>>(h2, lin_w, lin_b, (float*)d_out);
}
